// BilinearInteraction_49778670961367
// MI455X (gfx1250) — matrix operand order repaired and checked by reading, not compiled, not run on hardware
//
#include <hip/hip_runtime.h>
#include <hip/hip_bf16.h>

typedef _Float16 v4h  __attribute__((ext_vector_type(4)));
typedef _Float16 v8h  __attribute__((ext_vector_type(8)));
typedef _Float16 v16h __attribute__((ext_vector_type(16)));
typedef float    v8f  __attribute__((ext_vector_type(8)));

// Problem constants (from reference: B=2048, F=24, E=128 -> P=276)
#define NB   2048
#define NF   24
#define NE   128
#define NP   276
#define LDK  136          // padded LDS row stride (f16 elems): 272B -> conflict-free frag loads
#define MT   128          // M (batch) tile per block

// Build a v16h fragment from two contiguous 16B runs in LDS.
__device__ __forceinline__ v16h ld_frag(const _Float16* __restrict__ row, int o0, int o1) {
    v8h a = *(const v8h*)(row + o0);
    v8h b = *(const v8h*)(row + o1);
    v16h r;
#pragma unroll
    for (int i = 0; i < 8; ++i) { r[i] = a[i]; r[i + 8] = b[i]; }
    return r;
}

__global__ __launch_bounds__(256)
void bilinear_pair_gemm(const float* __restrict__ x,   // [B, F, E]
                        const float* __restrict__ W,   // [P, E, E]
                        float*       __restrict__ out) // [B, P, E]
{
    extern __shared__ _Float16 smem[];
    _Float16* sAhi = smem;                 // [128][LDK]  A tile, hi part
    _Float16* sAlo = sAhi + MT * LDK;      // [128][LDK]  A tile, lo part
    _Float16* sWhi = sAlo + MT * LDK;      // [128][LDK]  W^T (n-major, k contiguous), hi
    _Float16* sWlo = sWhi + NE * LDK;      // [128][LDK]  W^T, lo

    const int t     = threadIdx.x;
    const int mtile = blockIdx.x;          // 0..15
    const int q     = blockIdx.y;          // pair index 0..275

    // Decode (fi, fj) from combinations(range(F), 2) linear index.
    int rem = q, fi = 0, span = NF - 1;
    while (rem >= span) { rem -= span; ++fi; --span; }
    const int fj = fi + 1 + rem;

    const int m0 = mtile * MT;             // global batch base of this tile

    // ---- Stage W_q transposed into LDS as f16 hi/lo (coalesced float4 reads) ----
    {
        const float4* Wp4 = (const float4*)(W + (size_t)q * NE * NE);
#pragma unroll
        for (int it = 0; it < (NE * NE / 4) / 256; ++it) {   // 16 iters
            int g4   = t + it * 256;
            float4 w = Wp4[g4];
            int base = g4 * 4;
            int k    = base >> 7;          // row of W (the e index)
            int n    = base & 127;         // col of W (the f index), 4 consecutive
            float v[4] = { w.x, w.y, w.z, w.w };
#pragma unroll
            for (int u = 0; u < 4; ++u) {
                _Float16 hi = (_Float16)v[u];
                _Float16 lo = (_Float16)(v[u] - (float)hi);
                sWhi[(n + u) * LDK + k] = hi;
                sWlo[(n + u) * LDK + k] = lo;
            }
        }
    }
    // ---- Stage A tile x[m0:m0+128, fi, :] into LDS as f16 hi/lo ----
    {
#pragma unroll
        for (int it = 0; it < (MT * NE / 4) / 256; ++it) {   // 16 iters
            int g4 = t + it * 256;
            int m  = g4 >> 5;              // row within tile
            int k  = (g4 & 31) * 4;        // 4 consecutive k
            const float4 a = *(const float4*)(x + (size_t)((m0 + m) * NF + fi) * NE + k);
            float v[4] = { a.x, a.y, a.z, a.w };
            v4h hi4, lo4;
#pragma unroll
            for (int u = 0; u < 4; ++u) {
                _Float16 hi = (_Float16)v[u];
                hi4[u] = hi;
                lo4[u] = (_Float16)(v[u] - (float)hi);
            }
            *(v4h*)(sAhi + m * LDK + k) = hi4;
            *(v4h*)(sAlo + m * LDK + k) = lo4;
        }
    }
    __syncthreads();

    // ---- Compute: wave wv owns rows [wv*16, wv*16+16), all 8 n-tiles ----
    const int wv     = t >> 5;
    const int lane   = t & 31;
    const int l16    = lane & 15;
    const int hiHalf = (lane >> 4) & 1;    // lanes 16..31
    const int rowA   = wv * 16 + l16;      // A row (M) this lane supplies

    v8f acc[8];
#pragma unroll
    for (int nt = 0; nt < 8; ++nt)
#pragma unroll
        for (int i = 0; i < 8; ++i) acc[nt][i] = 0.0f;

#pragma unroll
    for (int ks = 0; ks < 4; ++ks) {
        const int k0    = ks * 32;
        const int koffA = k0 + (hiHalf ? 8 : 0);    // A: e0-7 -> K+0..7, e8-15 -> K+16..23
        const int koffB = k0 + (hiHalf ? 8 : 0);    // B: e0-7 -> K+0..7, e8-15 -> K+16..23

        const v16h ahi = ld_frag(sAhi + rowA * LDK, koffA, koffA + 16);
        const v16h alo = ld_frag(sAlo + rowA * LDK, koffA, koffA + 16);

#pragma unroll
        for (int nt = 0; nt < 8; ++nt) {
            const int col = nt * 16 + l16;          // B column (N)
            const v16h bhi = ld_frag(sWhi + col * LDK, koffB, koffB + 16);
            const v16h blo = ld_frag(sWlo + col * LDK, koffB, koffB + 16);
            // Split-precision f32 emulation: hi*hi + hi*lo + lo*hi (drop lo*lo)
            acc[nt] = __builtin_amdgcn_wmma_f32_16x16x32_f16(
                false, ahi, false, bhi, (short)0, acc[nt], false, false);
            acc[nt] = __builtin_amdgcn_wmma_f32_16x16x32_f16(
                false, ahi, false, blo, (short)0, acc[nt], false, false);
            acc[nt] = __builtin_amdgcn_wmma_f32_16x16x32_f16(
                false, alo, false, bhi, (short)0, acc[nt], false, false);
        }
    }

    // ---- Epilogue: multiply by x[:, fj, :] and stream out (non-temporal) ----
    // C/D layout: lane l16 -> N, vgpr r -> M = r + 8*hiHalf.
#pragma unroll
    for (int nt = 0; nt < 8; ++nt) {
        const int n = nt * 16 + l16;
#pragma unroll
        for (int r = 0; r < 8; ++r) {
            const int m  = m0 + wv * 16 + hiHalf * 8 + r;
            const float xj = x[(size_t)(m * NF + fj) * NE + n];
            const float v  = acc[nt][r] * xj;
            __builtin_nontemporal_store(v, out + ((size_t)m * NP + q) * NE + n);
        }
    }
}

extern "C" void kernel_launch(void* const* d_in, const int* in_sizes, int n_in,
                              void* d_out, int out_size, void* d_ws, size_t ws_size,
                              hipStream_t stream) {
    const float* x = (const float*)d_in[0];   // [2048, 24, 128] f32
    const float* W = (const float*)d_in[1];   // [276, 128, 128] f32
    float* out = (float*)d_out;               // [2048, 276, 128] f32

    dim3 grid(NB / MT, NP, 1);                // (16, 276)
    dim3 block(256, 1, 1);                    // 8 wave32
    size_t lds_bytes = (size_t)4 * MT * LDK * sizeof(_Float16); // 139,264 B (<320 KB/WGP)
    bilinear_pair_gemm<<<grid, block, lds_bytes, stream>>>(x, W, out);
}
